// AttentionBlock_65824668778896
// MI455X (gfx1250) — matrix operand order repaired and checked by reading, not compiled, not run on hardware
//
#include <hip/hip_runtime.h>

// ---------------------------------------------------------------------------
// AttentionBlock for MI455X (gfx1250, wave32, WMMA 16x16x32 f16, async->LDS)
// B=4, C=256, H=W=64, N=4096, heads=4, hd=64, groups=8
// ---------------------------------------------------------------------------

typedef __attribute__((ext_vector_type(16))) _Float16 v16h;
typedef __attribute__((ext_vector_type(8)))  float    v8f;
typedef __attribute__((ext_vector_type(4)))  unsigned int u32x4;
typedef int v4i __attribute__((vector_size(16)));   // matches builtin param type

union HOp { v16h v; u32x4 u[2]; };

#define WMMA_F16(A, B, C) \
  __builtin_amdgcn_wmma_f32_16x16x32_f16(false, (A), false, (B), (short)0, (C), false, false)

// Async copy global->LDS, 16B per lane, tracked by ASYNCcnt.
// Signature (from clang diagnostic): (v4i AS1*, v4i AS3*, imm offset, imm cpol).
// AS1 generic->global is identity; AS3 offset is bits[31:0] of the generic
// LDS address.
#define ASYNC_B128(GP, LP)                                                    \
  __builtin_amdgcn_global_load_async_to_lds_b128(                             \
      (__attribute__((address_space(1))) v4i*)(uintptr_t)(GP),                \
      (__attribute__((address_space(3))) v4i*)(unsigned int)(uintptr_t)(LP),  \
      0, 0)

#define LOG2E 1.44269504088896340f

static __device__ __forceinline__ v8f v8zero() {
  v8f z = {0.f, 0.f, 0.f, 0.f, 0.f, 0.f, 0.f, 0.f};
  return z;
}

// ---------------------------------------------------------------------------
// Kernel 1: GroupNorm statistics. One block per (batch, group).
// Group = 32 channels x 4096 spatial = 131072 contiguous floats.
// ---------------------------------------------------------------------------
__global__ __launch_bounds__(256) void gn_stats_kernel(const float* __restrict__ x,
                                                       float* __restrict__ stats) {
  const int bg = blockIdx.x;                       // b*8 + g
  const float* base = x + (size_t)bg * 131072;
  float s = 0.f, sq = 0.f;
  for (int i = threadIdx.x * 4; i < 131072; i += 256 * 4) {
    float4 v = *(const float4*)(base + i);
    s += v.x + v.y + v.z + v.w;
    sq += v.x * v.x + v.y * v.y + v.z * v.z + v.w * v.w;
  }
  __shared__ float rs[256], rq[256];
  rs[threadIdx.x] = s; rq[threadIdx.x] = sq;
  __syncthreads();
  for (int off = 128; off > 0; off >>= 1) {
    if ((int)threadIdx.x < off) {
      rs[threadIdx.x] += rs[threadIdx.x + off];
      rq[threadIdx.x] += rq[threadIdx.x + off];
    }
    __syncthreads();
  }
  if (threadIdx.x == 0) {
    float mean = rs[0] * (1.0f / 131072.0f);
    float var  = rq[0] * (1.0f / 131072.0f) - mean * mean;
    stats[bg * 2]     = mean;
    stats[bg * 2 + 1] = rsqrtf(var + 1e-5f);
  }
}

// ---------------------------------------------------------------------------
// Kernel 2: convert both weight matrices to f16 (262144 elems, grid 1024).
// ---------------------------------------------------------------------------
__global__ __launch_bounds__(256) void cvt_w_kernel(const float* __restrict__ wq,
                                                    const float* __restrict__ wp,
                                                    _Float16* __restrict__ wqh,
                                                    _Float16* __restrict__ wph) {
  int i = blockIdx.x * 256 + threadIdx.x;
  if (i < 196608) wqh[i] = (_Float16)wq[i];
  else            wph[i - 196608] = (_Float16)wp[i - 196608];
}

// ---------------------------------------------------------------------------
// Kernel 3: normalize + transpose: x (B,C,N) f32 -> ht (B,N,C) f16.
// Block = 32-channel x 64-token tile via LDS. Grid = 4 * 8 * 64 = 2048.
// ---------------------------------------------------------------------------
__global__ __launch_bounds__(256) void norm_kernel(const float* __restrict__ x,
                                                   const float* __restrict__ stats,
                                                   const float* __restrict__ gamma,
                                                   const float* __restrict__ beta,
                                                   _Float16* __restrict__ ht) {
  const int bid = blockIdx.x;
  const int b  = bid >> 9;
  const int ct = (bid >> 6) & 7;
  const int nt = bid & 63;
  const int cbase = ct * 32, nbase = nt * 64;
  __shared__ float tile[32][65];
  const int t = threadIdx.x;
#pragma unroll
  for (int i = 0; i < 8; ++i) {
    int idx = t + i * 256;
    int c = idx >> 6, n = idx & 63;
    int ch = cbase + c;
    int g = ch >> 5;
    float mean = stats[(b * 8 + g) * 2];
    float rstd = stats[(b * 8 + g) * 2 + 1];
    float v = x[((size_t)(b * 256 + ch)) * 4096 + nbase + n];
    tile[c][n] = (v - mean) * rstd * gamma[ch] + beta[ch];
  }
  __syncthreads();
#pragma unroll
  for (int i = 0; i < 8; ++i) {
    int idx = t + i * 256;
    int n = idx >> 5, c = idx & 31;
    ht[((size_t)(b * 4096 + nbase + n)) * 256 + cbase + c] = (_Float16)tile[c][n];
  }
}

// ---------------------------------------------------------------------------
// Kernel 4: QKV GEMM.  qkvT[j][tok] = sum_c wqkv[j][c] * ht[tok][c] + b[j]
// Each wave: 32 j x 64 tokens (2 A-tiles x 4 B-tiles = 8 WMMA / k-step).
// Outputs: q,k (B,nh,N,hd) f16 (q pre-scaled by 0.125), vt (B,nh,hd,N) f16.
// Grid = 4 batches * 24 jtiles * 64 ttiles / 8 waves = 768 blocks.
// ---------------------------------------------------------------------------
__global__ __launch_bounds__(256) void qkv_kernel(const _Float16* __restrict__ ht,
                                                  const _Float16* __restrict__ wq,
                                                  const float* __restrict__ bqkv,
                                                  _Float16* __restrict__ q,
                                                  _Float16* __restrict__ k,
                                                  _Float16* __restrict__ vt) {
  const int wid  = blockIdx.x * 8 + (threadIdx.x >> 5);
  const int lane = threadIdx.x & 31;
  const int l16  = lane & 15;
  const int lh   = lane >> 4;
  const int b    = wid / 1536;           // 24*64 jobs per batch
  const int rem  = wid % 1536;
  const int jbase = (rem >> 6) * 32;
  const int tbase = (rem & 63) * 64;

  v8f acc[2][4];
#pragma unroll
  for (int a = 0; a < 2; ++a)
#pragma unroll
    for (int s = 0; s < 4; ++s) acc[a][s] = v8zero();

  const _Float16* arow0 = wq + (size_t)(jbase + l16) * 256 + lh * 8;
  const _Float16* arow1 = arow0 + 16 * 256;
  const _Float16* brow  = ht + ((size_t)(b * 4096 + tbase + l16)) * 256 + lh * 8;

#pragma unroll
  for (int kc = 0; kc < 8; ++kc) {
    const int c = kc * 32;
    HOp a0, a1;
    a0.u[0] = *(const u32x4*)(arow0 + c);
    a0.u[1] = *(const u32x4*)(arow0 + c + 16);
    a1.u[0] = *(const u32x4*)(arow1 + c);
    a1.u[1] = *(const u32x4*)(arow1 + c + 16);
#pragma unroll
    for (int s = 0; s < 4; ++s) {
      HOp bb;
      const _Float16* p = brow + (size_t)s * 16 * 256 + c;
      bb.u[0] = *(const u32x4*)(p);
      bb.u[1] = *(const u32x4*)(p + 16);
      acc[0][s] = WMMA_F16(a0.v, bb.v, acc[0][s]);
      acc[1][s] = WMMA_F16(a1.v, bb.v, acc[1][s]);
    }
  }

#pragma unroll
  for (int a = 0; a < 2; ++a) {
#pragma unroll
    for (int r = 0; r < 8; ++r) {
      const int j = jbase + a * 16 + r + 8 * lh;
      const float bj = bqkv[j];
#pragma unroll
      for (int s = 0; s < 4; ++s) {
        const int tok = tbase + s * 16 + l16;
        const float val = acc[a][s][r] + bj;
        if (jbase < 256) {        // Q (fold in softmax scale 1/sqrt(64))
          int head = j >> 6, dim = j & 63;
          q[(((size_t)(b * 4 + head)) * 4096 + tok) * 64 + dim] = (_Float16)(val * 0.125f);
        } else if (jbase < 512) { // K
          int j2 = j - 256; int head = j2 >> 6, dim = j2 & 63;
          k[(((size_t)(b * 4 + head)) * 4096 + tok) * 64 + dim] = (_Float16)val;
        } else {                  // V, stored transposed (hd, N)
          int j2 = j - 512; int head = j2 >> 6, dim = j2 & 63;
          vt[(((size_t)(b * 4 + head)) * 64 + dim) * 4096 + tok] = (_Float16)val;
        }
      }
    }
  }
}

// ---------------------------------------------------------------------------
// Kernel 5: flash attention. One wave = 16 queries, 128 chunks of 32 keys.
// S^T = K @ Q^T so P lands directly in the A-operand register layout for
// O = P @ V. Block = 8 waves sharing double-buffered K / V^T tiles in LDS,
// filled with GLOBAL_LOAD_ASYNC_TO_LDS_B128 (ASYNCcnt), prefetched one
// chunk ahead. Grid = B*nh*(N/128) = 512 blocks.
// ---------------------------------------------------------------------------
__global__ __launch_bounds__(256) void flash_kernel(const _Float16* __restrict__ q,
                                                    const _Float16* __restrict__ k,
                                                    const _Float16* __restrict__ vt,
                                                    _Float16* __restrict__ ao) {
  const int tid  = threadIdx.x;
  const int wave = tid >> 5;
  const int lane = tid & 31;
  const int l16  = lane & 15;
  const int lh   = lane >> 4;
  const int bh   = blockIdx.x >> 5;   // b*4 + head
  const int qblk = blockIdx.x & 31;

  const _Float16* qb = q  + ((size_t)bh * 4096) * 64;
  const _Float16* kb = k  + ((size_t)bh * 4096) * 64;
  const _Float16* vb = vt + ((size_t)bh * 64) * 4096;

  __shared__ alignas(16) _Float16 sk2[2][32 * 72];   // 32 keys x 64 dims (+pad)
  __shared__ alignas(16) _Float16 sv2[2][64 * 40];   // 64 dims x 32 keys (+pad)

  // Q as B-operand for S^T = K @ Q^T (2 k-steps of 32 dims)
  const int qrow = qblk * 128 + wave * 16 + l16;
  HOp qop0, qop1;
  {
    const _Float16* qp = qb + (size_t)qrow * 64 + lh * 8;
    qop0.u[0] = *(const u32x4*)(qp);      qop0.u[1] = *(const u32x4*)(qp + 16);
    qop1.u[0] = *(const u32x4*)(qp + 32); qop1.u[1] = *(const u32x4*)(qp + 48);
  }

  // cooperative async fills: 256 threads x 16B per buffer
  const int krow = tid >> 3, kcol = (tid & 7) * 8;   // K chunk 32x64
  const int vrow = tid >> 2, vcol = (tid & 3) * 8;   // V^T chunk 64x32
  const _Float16* kg = kb + (size_t)krow * 64 + kcol;    // +2048 elems / chunk
  const _Float16* vg = vb + (size_t)vrow * 4096 + vcol;  // +32 elems / chunk

  // prefetch chunk 0 into buffer 0 (2 async ops per wave)
  ASYNC_B128(kg, &sk2[0][krow * 72 + kcol]);
  ASYNC_B128(vg, &sv2[0][vrow * 40 + vcol]);

  v8f o0 = v8zero(), o1 = v8zero(), o2 = v8zero(), o3 = v8zero();
  float mrun = -1e30f, lrun = 0.f;

  for (int c = 0; c < 128; ++c) {
    if (c + 1 < 128) {   // issue next chunk into ping buffer, then wait for
      const int nb = (c + 1) & 1;                //  the 2 ops of chunk c
      ASYNC_B128(kg + (size_t)(c + 1) * 2048, &sk2[nb][krow * 72 + kcol]);
      ASYNC_B128(vg + (size_t)(c + 1) * 32,   &sv2[nb][vrow * 40 + vcol]);
      asm volatile("s_wait_asynccnt 0x2" ::: "memory");
    } else {
      asm volatile("s_wait_asynccnt 0x0" ::: "memory");
    }
    __syncthreads();
    const _Float16* skc = sk2[c & 1];
    const _Float16* svc = sv2[c & 1];

    // S^T: two 16-key tiles, K-dim = 64 (2 WMMA each)
    v8f st0 = v8zero(), st1 = v8zero();
#pragma unroll
    for (int ks = 0; ks < 2; ++ks) {
      HOp a;
      const _Float16* p = &skc[(0 * 16 + l16) * 72 + ks * 32 + lh * 8];
      a.u[0] = *(const u32x4*)p; a.u[1] = *(const u32x4*)(p + 16);
      st0 = WMMA_F16(a.v, ks ? qop1.v : qop0.v, st0);
    }
#pragma unroll
    for (int ks = 0; ks < 2; ++ks) {
      HOp a;
      const _Float16* p = &skc[(1 * 16 + l16) * 72 + ks * 32 + lh * 8];
      a.u[0] = *(const u32x4*)p; a.u[1] = *(const u32x4*)(p + 16);
      st1 = WMMA_F16(a.v, ks ? qop1.v : qop0.v, st1);
    }

    // online softmax (lane-local per query, merge halves with one xor-shuffle)
    float mnew = mrun;
#pragma unroll
    for (int r = 0; r < 8; ++r) { mnew = fmaxf(mnew, st0[r]); mnew = fmaxf(mnew, st1[r]); }
    mnew = fmaxf(mnew, __shfl_xor(mnew, 16));
    const float alpha = exp2f((mrun - mnew) * LOG2E);
    mrun = mnew;

    HOp pa;
    float lsum = 0.f;
#pragma unroll
    for (int r = 0; r < 8; ++r) {
      float p0 = exp2f((st0[r] - mnew) * LOG2E);
      float p1 = exp2f((st1[r] - mnew) * LOG2E);
      lsum += p0 + p1;
      pa.v[r]     = (_Float16)p0;   // keys  8*lh + r     -> A elems 0..7
      pa.v[8 + r] = (_Float16)p1;   // keys 16 + 8*lh + r -> A elems 8..15
    }
    lrun = lrun * alpha + lsum;

    // rescale O rows (row = r + 8*lh; alpha lives in lane (query))
#pragma unroll
    for (int r = 0; r < 8; ++r) {
      const float ar = __shfl(alpha, r + 8 * lh);
      o0[r] *= ar; o1[r] *= ar; o2[r] *= ar; o3[r] *= ar;
    }

    // O += P @ V  (A = pa registers, B = V columns from V^T rows in LDS)
    {
      HOp bv;
      const _Float16* p = &svc[(0 * 16 + l16) * 40 + lh * 8];
      bv.u[0] = *(const u32x4*)p; bv.u[1] = *(const u32x4*)(p + 16);
      o0 = WMMA_F16(pa.v, bv.v, o0);
      p = &svc[(1 * 16 + l16) * 40 + lh * 8];
      bv.u[0] = *(const u32x4*)p; bv.u[1] = *(const u32x4*)(p + 16);
      o1 = WMMA_F16(pa.v, bv.v, o1);
      p = &svc[(2 * 16 + l16) * 40 + lh * 8];
      bv.u[0] = *(const u32x4*)p; bv.u[1] = *(const u32x4*)(p + 16);
      o2 = WMMA_F16(pa.v, bv.v, o2);
      p = &svc[(3 * 16 + l16) * 40 + lh * 8];
      bv.u[0] = *(const u32x4*)p; bv.u[1] = *(const u32x4*)(p + 16);
      o3 = WMMA_F16(pa.v, bv.v, o3);
    }
    __syncthreads();   // protect ping buffer before next-next async fill
  }

  // finalize: merge l across halves, divide, store to ao (B, N, C) f16
  const float ltot = lrun + __shfl_xor(lrun, 16);
  const float linv = 1.0f / ltot;
  const int bb = bh >> 2, h = bh & 3;
  v8f os[4] = {o0, o1, o2, o3};
#pragma unroll
  for (int r = 0; r < 8; ++r) {
    const float lr = __shfl(linv, r + 8 * lh);
    const int tok = qblk * 128 + wave * 16 + r + 8 * lh;
#pragma unroll
    for (int g = 0; g < 4; ++g) {
      const int ch = h * 64 + g * 16 + l16;
      ao[((size_t)(bb * 4096 + tok)) * 256 + ch] = (_Float16)(os[g][r] * lr);
    }
  }
}

// ---------------------------------------------------------------------------
// Kernel 6: output projection + bias + residual.
// out[b][j][n] = x[b][j][n] + sum_c ao[b][n][c]*wproj[j][c] + bproj[j]
// Wave = 32 j x 64 tokens.  Grid = 4*512/8 = 256 blocks.
// ---------------------------------------------------------------------------
__global__ __launch_bounds__(256) void proj_kernel(const _Float16* __restrict__ ao,
                                                   const _Float16* __restrict__ wp,
                                                   const float* __restrict__ bp,
                                                   const float* __restrict__ x,
                                                   float* __restrict__ out) {
  const int wid  = blockIdx.x * 8 + (threadIdx.x >> 5);
  const int lane = threadIdx.x & 31;
  const int l16  = lane & 15;
  const int lh   = lane >> 4;
  const int b    = wid >> 9;            // 8*64 = 512 jobs per batch
  const int rem  = wid & 511;
  const int jbase = (rem >> 6) * 32;
  const int tbase = (rem & 63) * 64;

  v8f acc[2][4];
#pragma unroll
  for (int a = 0; a < 2; ++a)
#pragma unroll
    for (int s = 0; s < 4; ++s) acc[a][s] = v8zero();

  const _Float16* arow0 = wp + (size_t)(jbase + l16) * 256 + lh * 8;
  const _Float16* arow1 = arow0 + 16 * 256;
  const _Float16* brow  = ao + ((size_t)(b * 4096 + tbase + l16)) * 256 + lh * 8;

#pragma unroll
  for (int kc = 0; kc < 8; ++kc) {
    const int c = kc * 32;
    HOp a0, a1;
    a0.u[0] = *(const u32x4*)(arow0 + c);
    a0.u[1] = *(const u32x4*)(arow0 + c + 16);
    a1.u[0] = *(const u32x4*)(arow1 + c);
    a1.u[1] = *(const u32x4*)(arow1 + c + 16);
#pragma unroll
    for (int s = 0; s < 4; ++s) {
      HOp bb;
      const _Float16* p = brow + (size_t)s * 16 * 256 + c;
      bb.u[0] = *(const u32x4*)(p);
      bb.u[1] = *(const u32x4*)(p + 16);
      acc[0][s] = WMMA_F16(a0.v, bb.v, acc[0][s]);
      acc[1][s] = WMMA_F16(a1.v, bb.v, acc[1][s]);
    }
  }

#pragma unroll
  for (int a = 0; a < 2; ++a) {
#pragma unroll
    for (int r = 0; r < 8; ++r) {
      const int j = jbase + a * 16 + r + 8 * lh;
      const float bj = bp[j];
#pragma unroll
      for (int s = 0; s < 4; ++s) {
        const int tok = tbase + s * 16 + l16;
        const size_t idx = ((size_t)(b * 256 + j)) * 4096 + tok;
        out[idx] = x[idx] + acc[a][s][r] + bj;
      }
    }
  }
}

// ---------------------------------------------------------------------------
extern "C" void kernel_launch(void* const* d_in, const int* in_sizes, int n_in,
                              void* d_out, int out_size, void* d_ws, size_t ws_size,
                              hipStream_t stream) {
  (void)in_sizes; (void)n_in; (void)out_size; (void)ws_size;
  const float* x     = (const float*)d_in[0];
  const float* gamma = (const float*)d_in[1];
  const float* beta  = (const float*)d_in[2];
  const float* wqkv  = (const float*)d_in[3];
  const float* bqkv  = (const float*)d_in[4];
  const float* wproj = (const float*)d_in[5];
  const float* bproj = (const float*)d_in[6];
  float* out = (float*)d_out;

  char* w = (char*)d_ws;
  float*    stats = (float*)w;     w += 256;
  _Float16* wq_h  = (_Float16*)w;  w += (size_t)768 * 256 * 2;
  _Float16* wp_h  = (_Float16*)w;  w += (size_t)256 * 256 * 2;
  _Float16* ht    = (_Float16*)w;  w += (size_t)4 * 4096 * 256 * 2;
  _Float16* qh    = (_Float16*)w;  w += (size_t)4 * 4 * 4096 * 64 * 2;
  _Float16* kh    = (_Float16*)w;  w += (size_t)4 * 4 * 4096 * 64 * 2;
  _Float16* vth   = (_Float16*)w;  w += (size_t)4 * 4 * 4096 * 64 * 2;
  _Float16* ao    = (_Float16*)w;

  gn_stats_kernel<<<32, 256, 0, stream>>>(x, stats);
  cvt_w_kernel<<<1024, 256, 0, stream>>>(wqkv, wproj, wq_h, wp_h);
  norm_kernel<<<2048, 256, 0, stream>>>(x, stats, gamma, beta, ht);
  qkv_kernel<<<768, 256, 0, stream>>>(ht, wq_h, bqkv, qh, kh, vth);
  flash_kernel<<<512, 256, 0, stream>>>(qh, kh, vth, ao);
  proj_kernel<<<256, 256, 0, stream>>>(ao, wp_h, bproj, x, out);
}
